// StateLSTM_31095563223215
// MI455X (gfx1250) — matrix operand order repaired and checked by reading, not compiled, not run on hardware
//
#include <hip/hip_runtime.h>
#include <hip/hip_bf16.h>
#include <stdint.h>

// ---------------------------------------------------------------------------
// Problem constants (from reference)
// ---------------------------------------------------------------------------
constexpr int L   = 32768;  // SEQ_LEN
constexpr int IN  = 256;    // INPUT_SIZE
constexpr int H   = 256;    // HIDDEN_SIZE
constexpr int G4  = 1024;   // 4*H gates
constexpr int O   = 128;    // OUTPUT_SIZE

// Workspace layout (bytes)
constexpr size_t OFF_XB  = 0;                                   // x bf16      16 MB
constexpr size_t OFF_WIH = OFF_XB  + (size_t)L  * IN * 2;       // W_ih bf16  512 KB
constexpr size_t OFF_WHH = OFF_WIH + (size_t)G4 * IN * 2;       // W_hh bf16  512 KB
constexpr size_t OFF_WFC = OFF_WHH + (size_t)G4 * H  * 2;       // W_fc bf16   64 KB
constexpr size_t OFF_XG  = OFF_WFC + (size_t)O  * H  * 2;       // xg f32     128 MB
constexpr size_t OFF_HS  = OFF_XG  + (size_t)L  * G4 * 4;       // relu(h) bf16 16 MB

// ---------------------------------------------------------------------------
// Types
// ---------------------------------------------------------------------------
typedef __attribute__((ext_vector_type(16))) __bf16 v16bf;
typedef __attribute__((ext_vector_type(8)))  float  v8f;

union TileU { v16bf v; uint4 q[2]; };

static __device__ __forceinline__ v8f wmma_bf16(v16bf a, v16bf b, v8f c) {
  return __builtin_amdgcn_wmma_f32_16x16x32_bf16(
      /*neg_a=*/false, a, /*neg_b=*/false, b,
      /*c_mod=*/(short)0, c, /*reuse_a=*/false, /*reuse_b=*/false);
}

static __device__ __forceinline__ uint16_t f2bf(float f) {
  uint32_t u = __float_as_uint(f);
  u += 0x7FFFu + ((u >> 16) & 1u);   // round-to-nearest-even
  return (uint16_t)(u >> 16);
}

// A-tile (16x32 bf16, MxK) from row-major bf16 matrix [rows x ld].
// ISA layout: lanes 0-15 hold row M=lane, K = 0-7 (v0-3) and 16-23 (v4-7);
// lanes 16-31 hold K = 8-15 and 24-31.
static __device__ __forceinline__ v16bf loadA(const uint16_t* base, int ld,
                                              int row_base, int kb, int lane) {
  int m   = row_base + (lane & 15);
  int sel = (lane >> 4) & 1;
  const uint16_t* p = base + (size_t)m * ld + kb + sel * 8;
  TileU t;
  t.q[0] = *reinterpret_cast<const uint4*>(p);        // K = kb + sel*8 .. +7
  t.q[1] = *reinterpret_cast<const uint4*>(p + 16);   // K = kb+16+sel*8 .. +7
  return t.v;
}

// A-tile from LDS h-vector; all 16 "rows" replicated (only D row 0 consumed).
static __device__ __forceinline__ v16bf loadA_lds(const uint16_t* hbuf,
                                                  int kb, int lane) {
  int sel = (lane >> 4) & 1;
  const uint16_t* p = hbuf + kb + sel * 8;
  TileU t;
  t.q[0] = *reinterpret_cast<const uint4*>(p);
  t.q[1] = *reinterpret_cast<const uint4*>(p + 16);
  return t.v;
}

// B-tile (32x16 bf16, KxN) from an N-major weight matrix W[n][k] (row-major NxK),
// i.e. B = W^T tile. ISA layout: lanes 0-15 hold N=lane%16, K=0-7,16-23; lanes 16-31 K=8-15,24-31.
static __device__ __forceinline__ v16bf loadB(const uint16_t* base, int ld,
                                              int n_base, int kb, int lane) {
  int n   = n_base + (lane & 15);
  int sel = (lane >> 4) & 1;
  const uint16_t* p = base + (size_t)n * ld + kb + sel * 8;
  TileU t;
  t.q[0] = *reinterpret_cast<const uint4*>(p);        // K = kb+sel*8 .. +7
  t.q[1] = *reinterpret_cast<const uint4*>(p + 16);   // K = kb+16+sel*8 .. +7
  return t.v;
}

// ---------------------------------------------------------------------------
// Kernel 0: fp32 -> bf16 conversion
// ---------------------------------------------------------------------------
__global__ void cvt_bf16_kernel(const float* __restrict__ src,
                                uint16_t* __restrict__ dst, int n) {
  int i = blockIdx.x * blockDim.x + threadIdx.x;
  if (i < n) dst[i] = f2bf(src[i]);
}

// ---------------------------------------------------------------------------
// Kernel 1: xg = x @ W_ih^T + (b_ih + b_hh)    (L x G4)
// One wave per 16-row M-strip; loops over all 64 N-tiles, A held in registers.
// ---------------------------------------------------------------------------
__global__ void xg_gemm_kernel(const uint16_t* __restrict__ xb,
                               const uint16_t* __restrict__ wih,
                               const float* __restrict__ b_ih,
                               const float* __restrict__ b_hh,
                               float* __restrict__ xg) {
  int wave = (blockIdx.x * blockDim.x + threadIdx.x) >> 5;  // 0..2047
  int lane = threadIdx.x & 31;
  int mb   = wave * 16;

  v16bf At[8];
#pragma unroll
  for (int kt = 0; kt < 8; ++kt) At[kt] = loadA(xb, IN, mb, kt * 32, lane);

  int rowoff = ((lane >> 4) & 1) * 8;
  for (int nt = 0; nt < G4 / 16; ++nt) {
    int nb  = nt * 16;
    v8f acc = {};
#pragma unroll
    for (int kt = 0; kt < 8; ++kt) {
      v16bf b = loadB(wih, IN, nb, kt * 32, lane);
      acc = wmma_bf16(At[kt], b, acc);
    }
    int   n    = nb + (lane & 15);
    float bias = b_ih[n] + b_hh[n];
#pragma unroll
    for (int r = 0; r < 8; ++r)
      xg[(size_t)(mb + r + rowoff) * G4 + n] = acc[r] + bias;
  }
}

// ---------------------------------------------------------------------------
// Kernel 2: sequential LSTM recurrence. ONE workgroup (32 wave32s on one WGP).
// Each wave owns 32 gate rows of W_hh, held in registers as pre-swizzled
// WMMA B-tiles (2 N-tiles x 8 K-tiles x 8 VGPR = 128 VGPRs of weights).
// h lives in LDS as bf16. Per step: 16 WMMAs/wave split into 4 independent
// even/odd-K chains (halves the dependent-WMMA critical path), 2 barriers.
// ---------------------------------------------------------------------------
__global__ void __launch_bounds__(1024, 1)
lstm_rec_kernel(const float* __restrict__ xg,
                const uint16_t* __restrict__ whh,
                uint16_t* __restrict__ hs) {
  __shared__ __align__(32) uint16_t hbuf[H];   // current h, bf16
  __shared__ float gbuf[G4];                   // gate pre-activations

  const int tid   = threadIdx.x;
  const int wave  = tid >> 5;
  const int lane  = tid & 31;
  const int gbase = wave * 32;

  // Preload this wave's weight tiles (stay in VGPRs across the whole loop).
  v16bf Bt0[8], Bt1[8];
#pragma unroll
  for (int kt = 0; kt < 8; ++kt) {
    Bt0[kt] = loadB(whh, H, gbase,      kt * 32, lane);
    Bt1[kt] = loadB(whh, H, gbase + 16, kt * 32, lane);
  }

  if (tid < H) hbuf[tid] = 0;   // h0 = 0
  float c = 0.0f;               // c state, owned by threads 0..255
  __syncthreads();

  for (int t = 0; t < L; ++t) {
    // Stream this step's xg contribution early (overlaps WMMA chain),
    // and prefetch ~64KB ahead of the sequential xg stream.
    float xg0 = 0.0f, xg1 = 0.0f;
    if (lane < 16) {
      int g0 = gbase + lane;
      xg0 = xg[(size_t)t * G4 + g0];
      xg1 = xg[(size_t)t * G4 + g0 + 16];
      __builtin_prefetch(&xg[(size_t)(t + 16) * G4 + g0], 0, 1);
    }

    // g[gbase..gbase+31] = h @ W_hh^T : 4 independent WMMA chains (ILP=4).
    v8f a0e = {}, a0o = {}, a1e = {}, a1o = {};
#pragma unroll
    for (int kt = 0; kt < 8; kt += 2) {
      v16bf ae = loadA_lds(hbuf, kt * 32, lane);
      v16bf ao = loadA_lds(hbuf, (kt + 1) * 32, lane);
      a0e = wmma_bf16(ae, Bt0[kt],     a0e);
      a1e = wmma_bf16(ae, Bt1[kt],     a1e);
      a0o = wmma_bf16(ao, Bt0[kt + 1], a0o);
      a1o = wmma_bf16(ao, Bt1[kt + 1], a1o);
    }
    if (lane < 16) {               // row M=0 of D: VGPR0 of lanes 0-15
      int g0 = gbase + lane;
      gbuf[g0]      = a0e[0] + a0o[0] + xg0;
      gbuf[g0 + 16] = a1e[0] + a1o[0] + xg1;
    }
    __syncthreads();

    // Element-wise gate update: one hidden unit per thread (0..255).
    if (tid < H) {
      float gi = gbuf[tid];
      float gf = gbuf[H + tid];
      float gg = gbuf[2 * H + tid];
      float go = gbuf[3 * H + tid];
      float i = 1.0f / (1.0f + __expf(-gi));
      float f = 1.0f / (1.0f + __expf(-gf));
      float g = 1.0f - 2.0f / (__expf(2.0f * gg) + 1.0f);   // tanh
      float o = 1.0f / (1.0f + __expf(-go));
      c = f * c + i * g;
      float tc = 1.0f - 2.0f / (__expf(2.0f * c) + 1.0f);   // tanh(c)
      float h  = o * tc;
      hbuf[tid] = f2bf(h);                                  // recurrent h
      hs[(size_t)t * H + tid] = f2bf(fmaxf(h, 0.0f));       // relu(h) for FC
    }
    __syncthreads();
  }
}

// ---------------------------------------------------------------------------
// Kernel 3: out = relu( relu(h) @ W_fc^T + b_fc )   (L x O), f32 output
// ---------------------------------------------------------------------------
__global__ void fc_gemm_kernel(const uint16_t* __restrict__ hsb,
                               const uint16_t* __restrict__ wfc,
                               const float* __restrict__ b_fc,
                               float* __restrict__ out) {
  int wave = (blockIdx.x * blockDim.x + threadIdx.x) >> 5;  // 0..2047
  int lane = threadIdx.x & 31;
  int mb   = wave * 16;

  v16bf At[8];
#pragma unroll
  for (int kt = 0; kt < 8; ++kt) At[kt] = loadA(hsb, H, mb, kt * 32, lane);

  int rowoff = ((lane >> 4) & 1) * 8;
#pragma unroll
  for (int nt = 0; nt < O / 16; ++nt) {
    int nb  = nt * 16;
    v8f acc = {};
#pragma unroll
    for (int kt = 0; kt < 8; ++kt) {
      v16bf b = loadB(wfc, H, nb, kt * 32, lane);
      acc = wmma_bf16(At[kt], b, acc);
    }
    int   n    = nb + (lane & 15);
    float bias = b_fc[n];
#pragma unroll
    for (int r = 0; r < 8; ++r)
      out[(size_t)(mb + r + rowoff) * O + n] = fmaxf(acc[r] + bias, 0.0f);
  }
}

// ---------------------------------------------------------------------------
// Host launcher
// ---------------------------------------------------------------------------
extern "C" void kernel_launch(void* const* d_in, const int* in_sizes, int n_in,
                              void* d_out, int out_size, void* d_ws, size_t ws_size,
                              hipStream_t stream) {
  const float* x    = (const float*)d_in[0];
  const float* W_ih = (const float*)d_in[1];
  const float* W_hh = (const float*)d_in[2];
  const float* b_ih = (const float*)d_in[3];
  const float* b_hh = (const float*)d_in[4];
  const float* W_fc = (const float*)d_in[5];
  const float* b_fc = (const float*)d_in[6];

  char* ws = (char*)d_ws;
  uint16_t* xb   = (uint16_t*)(ws + OFF_XB);
  uint16_t* wih  = (uint16_t*)(ws + OFF_WIH);
  uint16_t* whh  = (uint16_t*)(ws + OFF_WHH);
  uint16_t* wfc  = (uint16_t*)(ws + OFF_WFC);
  float*    xg   = (float*)   (ws + OFF_XG);
  uint16_t* hsb  = (uint16_t*)(ws + OFF_HS);

  // bf16 conversions
  {
    int n = L * IN;
    cvt_bf16_kernel<<<(n + 255) / 256, 256, 0, stream>>>(x, xb, n);
    n = G4 * IN;
    cvt_bf16_kernel<<<(n + 255) / 256, 256, 0, stream>>>(W_ih, wih, n);
    n = G4 * H;
    cvt_bf16_kernel<<<(n + 255) / 256, 256, 0, stream>>>(W_hh, whh, n);
    n = O * H;
    cvt_bf16_kernel<<<(n + 255) / 256, 256, 0, stream>>>(W_fc, wfc, n);
  }

  // Phase 1: input projection (2048 waves = 256 blocks x 256 threads)
  xg_gemm_kernel<<<256, 256, 0, stream>>>(xb, wih, b_ih, b_hh, xg);

  // Phase 2: sequential recurrence, single WGP
  lstm_rec_kernel<<<1, 1024, 0, stream>>>(xg, whh, hsb);

  // Phase 3: output FC
  fc_gemm_kernel<<<256, 256, 0, stream>>>(hsb, wfc, b_fc, (float*)d_out);
}
